// HierarchicalMoE_70720931496138
// MI455X (gfx1250) — compile-verified, matrix operand order checked by reading, not run on hardware
//
#include <hip/hip_runtime.h>
#include <hip/hip_bf16.h>
#include <math.h>

// ---------------- problem constants ----------------
#define H_DIM   768
#define I_DIM   3072
#define B_SZ    2
#define S_LEN   512
#define T_TOK   (B_SZ * S_LEN)      // 1024 tokens
#define N_GRP   4
#define GS      (N_GRP * S_LEN)     // 2048 tokens per batch after stacking
#define MT      (B_SZ * GS)         // 4096 rows for attention GEMMs
#define N_HEADS 8
#define HEAD_D  96                  // 768 / 8

typedef __attribute__((ext_vector_type(16))) __bf16 v16bf;
typedef __attribute__((ext_vector_type(8)))  __bf16 v8bf;
typedef __attribute__((ext_vector_type(8)))  float  v8f;

typedef __attribute__((address_space(3))) char lds_char;

__device__ __forceinline__ unsigned short f2bf(float f) {
  unsigned int u = __float_as_uint(f);
  unsigned int r = u + 0x7FFFu + ((u >> 16) & 1u);   // round-to-nearest-even
  return (unsigned short)(r >> 16);
}

__device__ __forceinline__ v8f zero8() {
  v8f z = {0.f, 0.f, 0.f, 0.f, 0.f, 0.f, 0.f, 0.f};
  return z;
}

// AS3 pointer value == LDS byte offset within the wave's LDS allocation.
__device__ __forceinline__ unsigned lds_off(const void* p) {
  return (unsigned)(unsigned long long)(const lds_char*)p;
}

// Async copy 16B global -> LDS (ASYNCcnt-tracked async datapath).
__device__ __forceinline__ void async_b128(unsigned loff, const void* gaddr) {
  asm volatile("global_load_async_to_lds_b128 %0, %1, off"
               :: "v"(loff), "v"((unsigned long long)(size_t)gaddr)
               : "memory");
}
__device__ __forceinline__ void wait_async0() {
  asm volatile("s_wait_asynccnt 0x0" ::: "memory");
}
__device__ __forceinline__ void wait_async1() {
  asm volatile("s_wait_asynccnt 0x1" ::: "memory");
}

// Build a 16-bit WMMA fragment (A-layout, per ISA: lanes 0-15 hold K=kbase..kbase+7
// in v0..3 and K=kbase+16..kbase+23 in v4..7; p must point at row_start + kbase).
__device__ __forceinline__ v16bf load_frag(const unsigned short* p) {
  v8bf lo = *(const v8bf*)(p);
  v8bf hi = *(const v8bf*)(p + 16);
  v16bf r;
#pragma unroll
  for (int i = 0; i < 8; i++) { r[i] = lo[i]; r[i + 8] = hi[i]; }
  return r;
}

__device__ __forceinline__ v8f wmma_bf16(v16bf a, v16bf b, v8f c) {
  return __builtin_amdgcn_wmma_f32_16x16x32_bf16(
      /*neg_a=*/false, a, /*neg_b=*/false, b,
      /*c_mod=*/(short)0, c, /*reuse_a=*/false, /*reuse_b=*/false);
}

// ---------------- converts ----------------
__global__ void f2bf_kernel(const float* __restrict__ in,
                            unsigned short* __restrict__ out, size_t n) {
  size_t i = (size_t)blockIdx.x * blockDim.x + threadIdx.x;
  if (i < n) out[i] = f2bf(in[i]);
}

// in: [K,N] row-major f32  ->  out: [N,K] row-major bf16
__global__ void f2bf_t_kernel(const float* __restrict__ in,
                              unsigned short* __restrict__ out, int K, int N) {
  size_t i = (size_t)blockIdx.x * blockDim.x + threadIdx.x;
  size_t tot = (size_t)K * N;
  if (i >= tot) return;
  int k = (int)(i / N), n = (int)(i % N);
  out[(size_t)n * K + k] = f2bf(in[i]);
}

// ---------------- routing (one wave per token) ----------------
__global__ __launch_bounds__(32) void route_kernel(
    const float* __restrict__ X, const float* __restrict__ gate,
    float* __restrict__ comb, int E) {
  const int t = blockIdx.x;
  const float* x = X + (size_t)t * H_DIM;
  float s[8];
  for (int e = 0; e < E; e++) {
    const float* gr = gate + (size_t)e * H_DIM;
    float p = 0.f;
    for (int h = threadIdx.x; h < H_DIM; h += 32) p += x[h] * gr[h];
#pragma unroll
    for (int off = 16; off > 0; off >>= 1) p += __shfl_xor(p, off, 32);
    s[e] = p;
  }
  float mx = s[0];
  for (int e = 1; e < E; e++) mx = fmaxf(mx, s[e]);
  float sum = 0.f;
  for (int e = 0; e < E; e++) { s[e] = __expf(s[e] - mx); sum += s[e]; }
  float inv = 1.f / sum;
  float fl = 1.0f - 0.001f * (float)E;
  float s2 = 0.f;
  for (int e = 0; e < E; e++) { s[e] = s[e] * inv * fl + 0.001f; s2 += s[e]; }
  inv = 1.f / s2;
  for (int e = 0; e < E; e++) s[e] *= inv;
  int i0 = 0;
  for (int e = 1; e < E; e++) if (s[e] > s[i0]) i0 = e;
  int i1 = (i0 == 0) ? 1 : 0;
  for (int e = 0; e < E; e++) if (e != i0 && s[e] > s[i1]) i1 = e;
  float tot2 = s[i0] + s[i1];
  if (threadIdx.x == 0) {
    for (int e = 0; e < E; e++) {
      float wv = (e == i0) ? s[i0] / tot2 : ((e == i1) ? s[i1] / tot2 : 0.f);
      comb[(size_t)t * E + e] = wv;
    }
  }
}

// ---------------- bf16 WMMA GEMM with async-LDS B staging ----------------
// C[M,N] = act(A[M,K] @ Bt[N,K]^T + bias)
// Block = 8 waves -> 128(M) x 64(N); B tile (64x32) double-buffered in LDS,
// filled by global_load_async_to_lds_b128. Last K-step peeled so the hot
// loop has unconditional issue -> s_wait_asynccnt 1 -> barrier -> 4x WMMA.
__global__ __launch_bounds__(256) void gemm_bf16(
    const unsigned short* __restrict__ A,
    const unsigned short* __restrict__ Bt,
    const float* __restrict__ bias,
    float* __restrict__ Cf,
    unsigned short* __restrict__ Cbf,
    int M, int N, int K, int act) {
  const int tid  = threadIdx.x;
  const int lane = tid & 31;
  const int wave = tid >> 5;
  const int l16  = lane & 15;
  const int half = lane >> 4;
  const int m0 = blockIdx.y * 128 + wave * 16;
  const int n0 = blockIdx.x * 64;

  __shared__ __align__(16) unsigned short Bl[2][64 * 32];

  // async staging: thread -> one 16B chunk; row = tid/4 (N), chunk = tid%4 (K)
  const int br = tid >> 2;            // 0..63
  const int bc = (tid & 3) * 8;       // 0,8,16,24 (elements)
  const unsigned short* Bsrc = Bt + (size_t)(n0 + br) * K + bc;
  const unsigned lbuf0 = lds_off(&Bl[0][br * 32 + bc]);
  const unsigned lbuf1 = lds_off(&Bl[1][br * 32 + bc]);

  v8f acc[4];
#pragma unroll
  for (int i = 0; i < 4; i++) acc[i] = zero8();

  const unsigned short* Ap = A + (size_t)(m0 + l16) * K + half * 8;

  async_b128(lbuf0, Bsrc);            // prime buffer 0 (k0 = 0)
  v16bf a_cur = load_frag(Ap);        // prime A fragment

  int buf = 0;
  int k0 = 0;
  // ---- main pipelined loop: all K-steps except the last ----
  for (; k0 < K - 32; k0 += 32) {
    async_b128(buf ? lbuf0 : lbuf1, Bsrc + k0 + 32);  // prefetch next tile
    wait_async1();                                    // current tile landed
    __syncthreads();                                  // all waves see tile

    v16bf a_next = load_frag(Ap + k0 + 32);
    __builtin_prefetch(Ap + k0 + 64, 0, 0);

    const unsigned short* Bb = &Bl[buf][half * 8];
#pragma unroll
    for (int nb = 0; nb < 4; nb++) {
      v16bf b = load_frag(Bb + (nb * 16 + l16) * 32);
      acc[nb] = wmma_bf16(a_cur, b, acc[nb]);
    }
    a_cur = a_next;
    __syncthreads();                                  // done reading buf
    buf ^= 1;
  }
  // ---- peeled final K-step ----
  {
    wait_async0();
    __syncthreads();
    const unsigned short* Bb = &Bl[buf][half * 8];
#pragma unroll
    for (int nb = 0; nb < 4; nb++) {
      v16bf b = load_frag(Bb + (nb * 16 + l16) * 32);
      acc[nb] = wmma_bf16(a_cur, b, acc[nb]);
    }
  }

  const int rbase = m0 + half * 8;
#pragma unroll
  for (int nb = 0; nb < 4; nb++) {
    int col = n0 + nb * 16 + l16;
    float bv = bias ? bias[col] : 0.f;
#pragma unroll
    for (int v = 0; v < 8; v++) {
      float val = acc[nb][v] + bv;
      if (act) val = 0.5f * val * (1.0f + erff(val * 0.70710678118654752f));
      size_t idx = (size_t)(rbase + v) * N + col;
      if (Cf)  Cf[idx]  = val;
      if (Cbf) Cbf[idx] = f2bf(val);
    }
  }
}

// ---------------- LayerNorm + residual + combine-accumulate ----------------
__global__ __launch_bounds__(256) void ln_combine_kernel(
    const float* __restrict__ outf, const float* __restrict__ specf,
    const float* __restrict__ g, const float* __restrict__ bln,
    const float* __restrict__ comb, float* __restrict__ gout,
    int accum, int E, int e) {
  const int t = blockIdx.x;
  const float* sp = specf + (size_t)t * H_DIM;
  __shared__ float red[256];
  float s = 0.f;
  for (int h = threadIdx.x; h < H_DIM; h += 256) s += sp[h];
  red[threadIdx.x] = s; __syncthreads();
  for (int st = 128; st > 0; st >>= 1) {
    if (threadIdx.x < st) red[threadIdx.x] += red[threadIdx.x + st];
    __syncthreads();
  }
  float mu = red[0] / (float)H_DIM;
  __syncthreads();
  float vs = 0.f;
  for (int h = threadIdx.x; h < H_DIM; h += 256) {
    float d = sp[h] - mu; vs += d * d;
  }
  red[threadIdx.x] = vs; __syncthreads();
  for (int st = 128; st > 0; st >>= 1) {
    if (threadIdx.x < st) red[threadIdx.x] += red[threadIdx.x + st];
    __syncthreads();
  }
  float invsd = rsqrtf(red[0] / (float)H_DIM + 1e-5f);
  float wgt = comb[(size_t)t * E + e];
  for (int h = threadIdx.x; h < H_DIM; h += 256) {
    float sv  = (sp[h] - mu) * invsd * g[h] + bln[h];
    float tot = outf[(size_t)t * H_DIM + h] + sv;
    size_t idx = (size_t)t * H_DIM + h;
    gout[idx] = accum ? (gout[idx] + wgt * tot) : (wgt * tot);
  }
}

// ---------------- stack group outputs + bf16 ----------------
__global__ void build_xcat_kernel(const float* __restrict__ gout,
                                  unsigned short* __restrict__ xcat) {
  size_t idx = (size_t)blockIdx.x * blockDim.x + threadIdx.x;
  size_t tot = (size_t)MT * H_DIM;
  if (idx >= tot) return;
  size_t h = idx % H_DIM;
  size_t r = idx / H_DIM;
  size_t sS = r % S_LEN; r /= S_LEN;
  size_t gI = r % N_GRP; size_t b = r / N_GRP;
  xcat[idx] = f2bf(gout[gI * ((size_t)T_TOK * H_DIM) +
                        ((b * S_LEN + sS) * H_DIM) + h]);
}

// ---------------- flash attention: one wave per 16 queries ----------------
__global__ __launch_bounds__(32) void attn_kernel(
    const unsigned short* __restrict__ Qb,
    const unsigned short* __restrict__ Kb,
    const unsigned short* __restrict__ Vb,
    float* __restrict__ Of) {
  const int qt   = blockIdx.x * 16;
  const int h    = blockIdx.y;
  const int b    = blockIdx.z;
  const int lane = threadIdx.x;
  const int l16  = lane & 15;
  const int half = lane >> 4;
  const size_t hbase = (size_t)h * HEAD_D;
  const float scale = 0.1020620726f;   // 1/sqrt(96)

  // Q fragments (A-layout), K dim = 96 -> 3 chunks of 32
  v16bf qf[3];
  {
    const unsigned short* qrow =
        Qb + ((size_t)b * GS + qt + l16) * H_DIM + hbase + half * 8;
#pragma unroll
    for (int kc = 0; kc < 3; kc++) qf[kc] = load_frag(qrow + kc * 32);
  }

  v8f o[6];
#pragma unroll
  for (int i = 0; i < 6; i++) o[i] = zero8();
  float m[8], l[8];
#pragma unroll
  for (int v = 0; v < 8; v++) { m[v] = -1e30f; l[v] = 0.f; }

  __shared__ __align__(16) unsigned short Pl[16 * 32];   // P tile (bf16)
  __shared__ __align__(16) unsigned short Vl[32 * 96];   // V tile (bf16, 6KB)

  for (int kt = 0; kt < GS; kt += 32) {
    // async-stage V tile [32 keys x 96 dims] into LDS; overlaps with S+softmax
    {
      const unsigned short* vrow =
          Vb + ((size_t)b * GS + kt + lane) * H_DIM + hbase;
#pragma unroll
      for (int c = 0; c < 12; c++)
        async_b128(lds_off(&Vl[lane * 96 + c * 8]), vrow + c * 8);
    }

    // S = Q @ K^T for two 16-key sub-tiles
    v8f sacc[2];
    sacc[0] = zero8(); sacc[1] = zero8();
#pragma unroll
    for (int st = 0; st < 2; st++) {
      const unsigned short* krow =
          Kb + ((size_t)b * GS + kt + st * 16 + l16) * H_DIM + hbase + half * 8;
#pragma unroll
      for (int kc = 0; kc < 3; kc++) {
        v16bf kf = load_frag(krow + kc * 32);
        sacc[st] = wmma_bf16(qf[kc], kf, sacc[st]);
      }
    }
    // online softmax update (row = v + half*8, cols across 16-lane half-wave)
    float fsc[8];
#pragma unroll
    for (int v = 0; v < 8; v++) {
      float x0 = sacc[0][v] * scale, x1 = sacc[1][v] * scale;
      float rm = fmaxf(x0, x1);
#pragma unroll
      for (int off = 1; off < 16; off <<= 1) rm = fmaxf(rm, __shfl_xor(rm, off, 32));
      float mn = fmaxf(m[v], rm);
      float p0 = __expf(x0 - mn), p1 = __expf(x1 - mn);
      float rs = p0 + p1;
#pragma unroll
      for (int off = 1; off < 16; off <<= 1) rs += __shfl_xor(rs, off, 32);
      float f = __expf(m[v] - mn);
      l[v] = l[v] * f + rs;
      m[v] = mn;
      fsc[v] = f;
      sacc[0][v] = p0; sacc[1][v] = p1;
    }
#pragma unroll
    for (int nb = 0; nb < 6; nb++)
#pragma unroll
      for (int v = 0; v < 8; v++) o[nb][v] *= fsc[v];

    // stage P (C-layout) into LDS row-major [16 q][32 k]
#pragma unroll
    for (int v = 0; v < 8; v++) {
      int row = v + half * 8;
      Pl[row * 32 + l16]      = f2bf(sacc[0][v]);
      Pl[row * 32 + 16 + l16] = f2bf(sacc[1][v]);
    }
    __syncthreads();
    // reload P as an A-layout fragment (ds_load_b128 x2)
    v16bf pf = load_frag(&Pl[l16 * 32 + half * 8]);

    wait_async0();                 // V tile landed in LDS
    __syncthreads();

    // O += P @ V ; V fragments gathered from LDS (stride-96 rows)
    const __bf16* Vl16 = (const __bf16*)Vl;
#pragma unroll
    for (int nb = 0; nb < 6; nb++) {
      int n = nb * 16 + l16;
      v16bf vf;
#pragma unroll
      for (int j = 0; j < 8; j++) {
        vf[j]     = Vl16[(half * 8 + j) * 96 + n];
        vf[j + 8] = Vl16[(half * 8 + 16 + j) * 96 + n];
      }
      o[nb] = wmma_bf16(pf, vf, o[nb]);
    }
    __syncthreads();
  }

  // epilogue: normalize by l and store
#pragma unroll
  for (int nb = 0; nb < 6; nb++) {
    int n = nb * 16 + l16;
#pragma unroll
    for (int v = 0; v < 8; v++) {
      int row = v + half * 8;
      Of[((size_t)b * GS + qt + row) * H_DIM + hbase + n] = o[nb][v] / l[v];
    }
  }
}

// ---------------- mean over groups ----------------
__global__ void mean_kernel(const float* __restrict__ proj,
                            float* __restrict__ out) {
  size_t idx = (size_t)blockIdx.x * blockDim.x + threadIdx.x;
  size_t tot = (size_t)T_TOK * H_DIM;
  if (idx >= tot) return;
  size_t h = idx % H_DIM;
  size_t r = idx / H_DIM;
  size_t sS = r % S_LEN; size_t b = r / S_LEN;
  float acc = 0.f;
  for (int gI = 0; gI < N_GRP; gI++)
    acc += proj[((size_t)b * GS + (size_t)gI * S_LEN + sS) * H_DIM + h];
  out[idx] = acc * 0.25f;
}

// ---------------- host launcher ----------------
extern "C" void kernel_launch(void* const* d_in, const int* in_sizes, int n_in,
                              void* d_out, int out_size, void* d_ws, size_t ws_size,
                              hipStream_t stream) {
  (void)in_sizes; (void)n_in; (void)out_size; (void)ws_size;
  const float* X = (const float*)d_in[0];
  static const int ES[4] = {8, 4, 2, 2};

  char* w = (char*)d_ws;
  size_t off = 0;
  auto take = [&](size_t bytes) -> char* {
    char* p = w + off;
    off += (bytes + 255) & ~(size_t)255;
    return p;
  };

  unsigned short* Xbf = (unsigned short*)take((size_t)T_TOK * H_DIM * 2);
  float* comb[4];
  for (int g = 0; g < 4; g++) comb[g] = (float*)take((size_t)T_TOK * ES[g] * 4);
  unsigned short* W1t   = (unsigned short*)take((size_t)H_DIM * I_DIM * 2);
  unsigned short* W2t   = (unsigned short*)take((size_t)H_DIM * I_DIM * 2);
  unsigned short* Wst   = (unsigned short*)take((size_t)H_DIM * H_DIM * 2);
  unsigned short* hidbf = (unsigned short*)take((size_t)T_TOK * I_DIM * 2);
  float*          outf  = (float*)take((size_t)T_TOK * H_DIM * 4);
  unsigned short* outbf = (unsigned short*)take((size_t)T_TOK * H_DIM * 2);
  float*          specf = (float*)take((size_t)T_TOK * H_DIM * 4);
  float*          gout  = (float*)take((size_t)N_GRP * T_TOK * H_DIM * 4);
  unsigned short* xcat  = (unsigned short*)take((size_t)MT * H_DIM * 2);
  unsigned short* wbf   = (unsigned short*)take((size_t)H_DIM * H_DIM * 2);
  unsigned short* Qbf   = (unsigned short*)take((size_t)MT * H_DIM * 2);
  unsigned short* Kbf   = (unsigned short*)take((size_t)MT * H_DIM * 2);
  unsigned short* Vbf   = (unsigned short*)take((size_t)MT * H_DIM * 2);
  float*          attnf = (float*)take((size_t)MT * H_DIM * 4);
  unsigned short* attnbf= (unsigned short*)take((size_t)MT * H_DIM * 2);
  float*          projf = (float*)take((size_t)MT * H_DIM * 4);

  // 1. X -> bf16
  {
    size_t n = (size_t)T_TOK * H_DIM;
    f2bf_kernel<<<(unsigned)((n + 255) / 256), 256, 0, stream>>>(X, Xbf, n);
  }
  // 2. routing per group
  for (int g = 0; g < 4; g++) {
    const float* gate = (const float*)d_in[1 + g * 9 + 0];
    route_kernel<<<T_TOK, 32, 0, stream>>>(X, gate, comb[g], ES[g]);
  }
  // 3. experts (dense dispatch)
  for (int g = 0; g < 4; g++) {
    int base = 1 + g * 9;
    const float* W1 = (const float*)d_in[base + 1];
    const float* b1 = (const float*)d_in[base + 2];
    const float* W2 = (const float*)d_in[base + 3];
    const float* b2 = (const float*)d_in[base + 4];
    const float* Ws = (const float*)d_in[base + 5];
    const float* bs = (const float*)d_in[base + 6];
    const float* lg = (const float*)d_in[base + 7];
    const float* lb = (const float*)d_in[base + 8];
    float* go = gout + (size_t)g * T_TOK * H_DIM;
    for (int e = 0; e < ES[g]; e++) {
      size_t n1 = (size_t)H_DIM * I_DIM;
      size_t ns = (size_t)H_DIM * H_DIM;
      f2bf_t_kernel<<<(unsigned)((n1 + 255) / 256), 256, 0, stream>>>(
          W1 + (size_t)e * H_DIM * I_DIM, W1t, H_DIM, I_DIM);
      gemm_bf16<<<dim3(I_DIM / 64, T_TOK / 128), 256, 0, stream>>>(
          Xbf, W1t, b1 + (size_t)e * I_DIM, nullptr, hidbf,
          T_TOK, I_DIM, H_DIM, 1);
      f2bf_t_kernel<<<(unsigned)((n1 + 255) / 256), 256, 0, stream>>>(
          W2 + (size_t)e * I_DIM * H_DIM, W2t, I_DIM, H_DIM);
      gemm_bf16<<<dim3(H_DIM / 64, T_TOK / 128), 256, 0, stream>>>(
          hidbf, W2t, b2 + (size_t)e * H_DIM, outf, outbf,
          T_TOK, H_DIM, I_DIM, 0);
      f2bf_t_kernel<<<(unsigned)((ns + 255) / 256), 256, 0, stream>>>(
          Ws + (size_t)e * H_DIM * H_DIM, Wst, H_DIM, H_DIM);
      gemm_bf16<<<dim3(H_DIM / 64, T_TOK / 128), 256, 0, stream>>>(
          outbf, Wst, bs + (size_t)e * H_DIM, specf, nullptr,
          T_TOK, H_DIM, H_DIM, 0);
      ln_combine_kernel<<<T_TOK, 256, 0, stream>>>(
          outf, specf, lg + (size_t)e * H_DIM, lb + (size_t)e * H_DIM,
          comb[g], go, (e > 0) ? 1 : 0, ES[g], e);
    }
  }
  // 4. stack group outputs -> xcat (bf16)
  {
    size_t n = (size_t)MT * H_DIM;
    build_xcat_kernel<<<(unsigned)((n + 255) / 256), 256, 0, stream>>>(gout, xcat);
  }
  // 5. QKV projections (Wq is [H,H], x @ Wq^T -> B-operand already [N,K])
  const float* Wq = (const float*)d_in[37]; const float* bq = (const float*)d_in[38];
  const float* Wk = (const float*)d_in[39]; const float* bk = (const float*)d_in[40];
  const float* Wv = (const float*)d_in[41]; const float* bv = (const float*)d_in[42];
  const float* Wo = (const float*)d_in[43]; const float* bo = (const float*)d_in[44];
  size_t nw = (size_t)H_DIM * H_DIM;
  f2bf_kernel<<<(unsigned)((nw + 255) / 256), 256, 0, stream>>>(Wq, wbf, nw);
  gemm_bf16<<<dim3(H_DIM / 64, MT / 128), 256, 0, stream>>>(
      xcat, wbf, bq, nullptr, Qbf, MT, H_DIM, H_DIM, 0);
  f2bf_kernel<<<(unsigned)((nw + 255) / 256), 256, 0, stream>>>(Wk, wbf, nw);
  gemm_bf16<<<dim3(H_DIM / 64, MT / 128), 256, 0, stream>>>(
      xcat, wbf, bk, nullptr, Kbf, MT, H_DIM, H_DIM, 0);
  f2bf_kernel<<<(unsigned)((nw + 255) / 256), 256, 0, stream>>>(Wv, wbf, nw);
  gemm_bf16<<<dim3(H_DIM / 64, MT / 128), 256, 0, stream>>>(
      xcat, wbf, bv, nullptr, Vbf, MT, H_DIM, H_DIM, 0);
  // 6. attention
  attn_kernel<<<dim3(GS / 16, N_HEADS, B_SZ), 32, 0, stream>>>(Qbf, Kbf, Vbf, attnf);
  // 7-8. O projection
  {
    size_t n = (size_t)MT * H_DIM;
    f2bf_kernel<<<(unsigned)((n + 255) / 256), 256, 0, stream>>>(attnf, attnbf, n);
  }
  f2bf_kernel<<<(unsigned)((nw + 255) / 256), 256, 0, stream>>>(Wo, wbf, nw);
  gemm_bf16<<<dim3(H_DIM / 64, MT / 128), 256, 0, stream>>>(
      attnbf, wbf, bo, projf, nullptr, MT, H_DIM, H_DIM, 0);
  // 9. mean over groups -> output
  {
    size_t n = (size_t)T_TOK * H_DIM;
    mean_kernel<<<(unsigned)((n + 255) / 256), 256, 0, stream>>>(projf, (float*)d_out);
  }
}
